// SparseMoeBlock_50646254355314
// MI455X (gfx1250) — matrix operand order repaired and checked by reading, not compiled, not run on hardware
//
#include <hip/hip_runtime.h>

#define D 2048
#define I 1408
#define E 16
#define NTOK 8192
#define TM 128
#define TN 64
#define TK 32
#define PMAX 18432   // 16384 pairs + per-expert padding to 128, rounded up
#define NTILES 144   // PMAX / TM
#define A_STRIDE 40  // ushorts per A row (32 + pad), 16B aligned, bank-spread
#define B_STRIDE 20  // u32 per B column (16 + pad), 16B aligned, bank-spread

typedef __attribute__((ext_vector_type(16))) __bf16 bf16x16;
typedef __attribute__((ext_vector_type(8)))  float  f32x8;

union FragU { bf16x16 v; uint4 q[2]; };

__device__ __forceinline__ unsigned short f2bf(float f) {
  unsigned int u = __float_as_uint(f);
  u += 0x7FFFu + ((u >> 16) & 1u);   // round-to-nearest-even (epilogue only)
  return (unsigned short)(u >> 16);
}
__device__ __forceinline__ float bf2f(unsigned short b) {
  return __uint_as_float(((unsigned int)b) << 16);
}
// pack two floats -> two bf16 in one u32 (lo = a, hi = b): single v_perm_b32
__device__ __forceinline__ unsigned int f2bf_pk(float a, float b) {
  return __builtin_amdgcn_perm(__float_as_uint(b), __float_as_uint(a), 0x07060302u);
}

// CDNA5 async global->LDS DMA (ASYNCcnt-tracked), per CDNA5 ISA
__device__ __forceinline__ void async_ld_b128(unsigned int lds_off, const void* gaddr) {
  asm volatile("global_load_async_to_lds_b128 %0, %1, off"
               :: "v"(lds_off), "v"(gaddr) : "memory");
}
__device__ __forceinline__ void wait_async0() {
  asm volatile("s_wait_asynccnt 0x0" ::: "memory");
}
__device__ __forceinline__ unsigned int lds_off_of(const void* p) {
  return (unsigned int)(size_t)p;   // LDS aperture: addr[31:0] = LDS byte offset
}

// ---------------- kernel 0: workspace init -----------------------------------
__global__ void moe_init(int* __restrict__ pair_token, int* __restrict__ counts,
                         int* __restrict__ tile_expert, int* __restrict__ tile_base,
                         unsigned short* __restrict__ zeropad) {
  int i = blockIdx.x * blockDim.x + threadIdx.x;
  if (i < PMAX) pair_token[i] = -1;
  if (i < E) counts[i] = 0;
  if (i < NTILES) { tile_expert[i] = -1; tile_base[i] = 0; }
  if (i < D) zeropad[i] = 0;
}

// ---------------- kernel 1: router + x -> bf16 -------------------------------
__global__ __launch_bounds__(256) void moe_router(
    const float* __restrict__ x, const float* __restrict__ gate_w,
    unsigned short* __restrict__ xb, int* __restrict__ sel,
    float* __restrict__ rwv, int* __restrict__ counts) {
  __shared__ float gw[E * 128];
  const int tid = threadIdx.x;
  const int t = blockIdx.x * 256 + tid;
  const float* xr = x + (size_t)t * D;
  unsigned short* xbr = xb + (size_t)t * D;
  float acc[E];
#pragma unroll
  for (int e = 0; e < E; ++e) acc[e] = 0.f;
  for (int d0 = 0; d0 < D; d0 += 128) {
    __syncthreads();
    for (int i = tid; i < E * 128; i += 256) {
      int e = i >> 7, dd = i & 127;
      gw[i] = gate_w[(size_t)e * D + d0 + dd];
    }
    __syncthreads();
    for (int d = 0; d < 128; d += 8) {
      float4 a = *(const float4*)(xr + d0 + d);
      float4 b = *(const float4*)(xr + d0 + d + 4);
      float xv[8] = {a.x, a.y, a.z, a.w, b.x, b.y, b.z, b.w};
      uint4 pk;
      pk.x = f2bf_pk(xv[0], xv[1]);
      pk.y = f2bf_pk(xv[2], xv[3]);
      pk.z = f2bf_pk(xv[4], xv[5]);
      pk.w = f2bf_pk(xv[6], xv[7]);
      *(uint4*)(xbr + d0 + d) = pk;
#pragma unroll
      for (int j = 0; j < 8; ++j) {
#pragma unroll
        for (int e = 0; e < E; ++e)
          acc[e] = fmaf(xv[j], gw[e * 128 + d + j], acc[e]);
      }
    }
  }
  // top-2 (earliest index wins ties, like lax.top_k)
  int e0 = 0; float l0 = acc[0];
#pragma unroll
  for (int e = 1; e < E; ++e) if (acc[e] > l0) { l0 = acc[e]; e0 = e; }
  int e1 = -1; float l1 = -3.4e38f;
#pragma unroll
  for (int e = 0; e < E; ++e) if (e != e0 && acc[e] > l1) { l1 = acc[e]; e1 = e; }
  float p1 = __expf(l1 - l0);        // normalized top-2 softmax weights
  float inv = 1.f / (1.f + p1);
  sel[2 * t] = e0; sel[2 * t + 1] = e1;
  rwv[2 * t] = inv; rwv[2 * t + 1] = p1 * inv;
  atomicAdd(&counts[e0], 1);
  atomicAdd(&counts[e1], 1);
}

// ---------------- kernel 2: offsets / tile map (serial, tiny) ----------------
__global__ void moe_setup(const int* __restrict__ counts, int* __restrict__ cursor,
                          int* __restrict__ tile_expert, int* __restrict__ tile_base) {
  if (threadIdx.x == 0 && blockIdx.x == 0) {
    int off = 0, tile = 0;
    for (int e = 0; e < E; ++e) {
      cursor[e] = off;
      int nt = (counts[e] + TM - 1) / TM;
      for (int i = 0; i < nt; ++i) {
        tile_expert[tile] = e;
        tile_base[tile] = off + i * TM;
        ++tile;
      }
      off += nt * TM;
    }
  }
}

// ---------------- kernel 3: scatter pairs into padded slots ------------------
__global__ void moe_assign(const int* __restrict__ sel, int* __restrict__ cursor,
                           int* __restrict__ pair_token, int* __restrict__ pairidx) {
  int t = blockIdx.x * blockDim.x + threadIdx.x;
  if (t >= NTOK) return;
#pragma unroll
  for (int k = 0; k < 2; ++k) {
    int e = sel[2 * t + k];
    int pos = atomicAdd(&cursor[e], 1);
    pair_token[pos] = t;
    pairidx[2 * t + k] = pos;
  }
}

// ---------------- kernel 4: fused gate/up GEMM + SiLU, bf16 WMMA -------------
// block tile 128(M) x 64(N) x 32(K); double-buffered LDS; A tile via
// GLOBAL_LOAD_ASYNC_TO_LDS_B128 (ASYNCcnt); B tiles f32->bf16 via v_perm pack.
__global__ __launch_bounds__(256) void moe_gateup(
    const unsigned short* __restrict__ xb, const float* __restrict__ wg,
    const float* __restrict__ wu, const int* __restrict__ pair_token,
    const int* __restrict__ tile_expert, const int* __restrict__ tile_base,
    const unsigned short* __restrict__ zeropad, unsigned short* __restrict__ Hb) {
  const int e = tile_expert[blockIdx.x];
  if (e < 0) return;
  const int base = tile_base[blockIdx.x];
  const int n0 = blockIdx.y * TN;

  __shared__ __align__(16) unsigned short Ash[2][TM * A_STRIDE];
  __shared__ __align__(16) unsigned int   Bgs[2][TN * B_STRIDE];
  __shared__ __align__(16) unsigned int   Bus[2][TN * B_STRIDE];
  __shared__ int toks[TM];

  const int tid = threadIdx.x;
  const int lane = tid & 31;
  const int wid = tid >> 5;
  const int mbase = wid * 16;                 // wave's 16-row strip
  const int half = lane >> 4, l16 = lane & 15;

  if (tid < TM) toks[tid] = pair_token[base + tid];
  __syncthreads();

  // cooperative-load mapping
  const int arow = tid >> 1;                  // 0..127
  const int acolh = tid & 1;                  // which 16-element half of k
  const int kp = tid >> 4;                    // k-pair 0..15 (k = 2*kp)
  const int n4 = (tid & 15) * 4;              // n chunk of 4
  const int atok = toks[arow];

  const unsigned short* ap =
      ((atok < 0) ? zeropad : xb + (size_t)atok * D) + acolh * 16;
  const unsigned int ldsA[2] = {
      lds_off_of(&Ash[0][arow * A_STRIDE + acolh * 16]),
      lds_off_of(&Ash[1][arow * A_STRIDE + acolh * 16])};

  const float* wg_e = wg + (size_t)e * D * I;
  const float* wu_e = wu + (size_t)e * D * I;

  f32x8 accg[4] = {{0.f}, {0.f}, {0.f}, {0.f}};
  f32x8 accu[4] = {{0.f}, {0.f}, {0.f}, {0.f}};

  float4 gS0, gS1, uS0, uS1;

  auto stageA = [&](int k0, int b) {
    async_ld_b128(ldsA[b], ap + k0);          // k 0..7 of this half
    async_ld_b128(ldsA[b] + 16, ap + k0 + 8); // k 8..15 of this half
  };
  auto stageB = [&](int k0) {
    const float* pg = wg_e + (size_t)(k0 + 2 * kp) * I + n0 + n4;
    gS0 = ((const float4*)pg)[0];
    gS1 = *(const float4*)(pg + I);
    const float* pu = wu_e + (size_t)(k0 + 2 * kp) * I + n0 + n4;
    uS0 = ((const float4*)pu)[0];
    uS1 = *(const float4*)(pu + I);
    if (k0 + 2 * TK < D) {
      __builtin_prefetch(pg + 2 * (size_t)TK * I, 0, 1);
      __builtin_prefetch(pu + 2 * (size_t)TK * I, 0, 1);
    }
  };
  auto commitB = [&](int b) {
    Bgs[b][(n4 + 0) * B_STRIDE + kp] = f2bf_pk(gS0.x, gS1.x);
    Bgs[b][(n4 + 1) * B_STRIDE + kp] = f2bf_pk(gS0.y, gS1.y);
    Bgs[b][(n4 + 2) * B_STRIDE + kp] = f2bf_pk(gS0.z, gS1.z);
    Bgs[b][(n4 + 3) * B_STRIDE + kp] = f2bf_pk(gS0.w, gS1.w);
    Bus[b][(n4 + 0) * B_STRIDE + kp] = f2bf_pk(uS0.x, uS1.x);
    Bus[b][(n4 + 1) * B_STRIDE + kp] = f2bf_pk(uS0.y, uS1.y);
    Bus[b][(n4 + 2) * B_STRIDE + kp] = f2bf_pk(uS0.z, uS1.z);
    Bus[b][(n4 + 3) * B_STRIDE + kp] = f2bf_pk(uS0.w, uS1.w);
  };

  stageA(0, 0);
  stageB(0);
  commitB(0);
  wait_async0();
  __syncthreads();

  const int nIter = D / TK;
  for (int it = 0; it < nIter; ++it) {
    const int cur = it & 1;
    if (it + 1 < nIter) {
      stageA((it + 1) * TK, cur ^ 1);   // async DMA into next buffer
      stageB((it + 1) * TK);            // weight tile into registers
    }

    FragU af;
    const int aoff = (mbase + l16) * A_STRIDE + half * 16;
    af.q[0] = *(const uint4*)&Ash[cur][aoff];
    af.q[1] = *(const uint4*)&Ash[cur][aoff + 8];
#pragma unroll
    for (int c = 0; c < 4; ++c) {
      FragU bg, bu;
      const int boff = (c * 16 + l16) * B_STRIDE + half * 8;
      bg.q[0] = *(const uint4*)&Bgs[cur][boff];
      bg.q[1] = *(const uint4*)&Bgs[cur][boff + 4];
      bu.q[0] = *(const uint4*)&Bus[cur][boff];
      bu.q[1] = *(const uint4*)&Bus[cur][boff + 4];
      accg[c] = __builtin_amdgcn_wmma_f32_16x16x32_bf16(
          false, af.v, false, bg.v, (short)0, accg[c], false, false);
      accu[c] = __builtin_amdgcn_wmma_f32_16x16x32_bf16(
          false, af.v, false, bu.v, (short)0, accu[c], false, false);
    }
    if (it + 1 < nIter) commitB(cur ^ 1);
    wait_async0();
    __syncthreads();
  }

  // epilogue: H = silu(G) * U, bf16
  const int srow = base + mbase + half * 8;
#pragma unroll
  for (int c = 0; c < 4; ++c) {
    const int col = n0 + c * 16 + l16;
#pragma unroll
    for (int r = 0; r < 8; ++r) {
      float g = accg[c][r];
      float u = accu[c][r];
      float s = __fdividef(g, 1.f + __expf(-g));
      Hb[(size_t)(srow + r) * I + col] = f2bf(s * u);
    }
  }
}

// ---------------- kernel 5: down-proj GEMM, bf16 WMMA ------------------------
__global__ __launch_bounds__(256) void moe_down(
    const unsigned short* __restrict__ Hb, const float* __restrict__ wd,
    const int* __restrict__ tile_expert, const int* __restrict__ tile_base,
    unsigned short* __restrict__ Yb) {
  const int e = tile_expert[blockIdx.x];
  if (e < 0) return;
  const int base = tile_base[blockIdx.x];
  const int n0 = blockIdx.y * TN;

  __shared__ __align__(16) unsigned short Ash[2][TM * A_STRIDE];
  __shared__ __align__(16) unsigned int   Bs[2][TN * B_STRIDE];

  const int tid = threadIdx.x;
  const int lane = tid & 31;
  const int wid = tid >> 5;
  const int mbase = wid * 16;
  const int half = lane >> 4, l16 = lane & 15;

  const int arow = tid >> 1;
  const int acolh = tid & 1;
  const int kp = tid >> 4;
  const int n4 = (tid & 15) * 4;

  const float* wd_e = wd + (size_t)e * I * D;
  const unsigned short* arp = Hb + (size_t)(base + arow) * I + acolh * 16;
  const unsigned int ldsA[2] = {
      lds_off_of(&Ash[0][arow * A_STRIDE + acolh * 16]),
      lds_off_of(&Ash[1][arow * A_STRIDE + acolh * 16])};

  f32x8 acc[4] = {{0.f}, {0.f}, {0.f}, {0.f}};

  float4 bS0, bS1;

  auto stageA = [&](int k0, int b) {
    async_ld_b128(ldsA[b], arp + k0);
    async_ld_b128(ldsA[b] + 16, arp + k0 + 8);
  };
  auto stageB = [&](int k0) {
    const float* pb = wd_e + (size_t)(k0 + 2 * kp) * D + n0 + n4;
    bS0 = ((const float4*)pb)[0];
    bS1 = *(const float4*)(pb + D);
    if (k0 + 2 * TK < I) __builtin_prefetch(pb + 2 * (size_t)TK * D, 0, 1);
  };
  auto commitB = [&](int b) {
    Bs[b][(n4 + 0) * B_STRIDE + kp] = f2bf_pk(bS0.x, bS1.x);
    Bs[b][(n4 + 1) * B_STRIDE + kp] = f2bf_pk(bS0.y, bS1.y);
    Bs[b][(n4 + 2) * B_STRIDE + kp] = f2bf_pk(bS0.z, bS1.z);
    Bs[b][(n4 + 3) * B_STRIDE + kp] = f2bf_pk(bS0.w, bS1.w);
  };

  stageA(0, 0);
  stageB(0);
  commitB(0);
  wait_async0();
  __syncthreads();

  const int nIter = I / TK;
  for (int it = 0; it < nIter; ++it) {
    const int cur = it & 1;
    if (it + 1 < nIter) {
      stageA((it + 1) * TK, cur ^ 1);
      stageB((it + 1) * TK);
    }

    FragU af;
    const int aoff = (mbase + l16) * A_STRIDE + half * 16;
    af.q[0] = *(const uint4*)&Ash[cur][aoff];
    af.q[1] = *(const uint4*)&Ash[cur][aoff + 8];
#pragma unroll
    for (int c = 0; c < 4; ++c) {
      FragU bf;
      const int boff = (c * 16 + l16) * B_STRIDE + half * 8;
      bf.q[0] = *(const uint4*)&Bs[cur][boff];
      bf.q[1] = *(const uint4*)&Bs[cur][boff + 4];
      acc[c] = __builtin_amdgcn_wmma_f32_16x16x32_bf16(
          false, af.v, false, bf.v, (short)0, acc[c], false, false);
    }
    if (it + 1 < nIter) commitB(cur ^ 1);
    wait_async0();
    __syncthreads();
  }

  const int srow = base + mbase + half * 8;
#pragma unroll
  for (int c = 0; c < 4; ++c) {
    const int col = n0 + c * 16 + l16;
#pragma unroll
    for (int r = 0; r < 8; ++r)
      Yb[(size_t)(srow + r) * D + col] = f2bf(acc[c][r]);
  }
}

// ---------------- kernel 6: combine per token --------------------------------
__global__ __launch_bounds__(256) void moe_combine(
    const unsigned short* __restrict__ Yb, const int* __restrict__ pairidx,
    const float* __restrict__ rwv, float* __restrict__ out) {
  const int t = blockIdx.x;
  const int p0 = pairidx[2 * t], p1 = pairidx[2 * t + 1];
  const float w0 = rwv[2 * t], w1 = rwv[2 * t + 1];
  const int c = threadIdx.x * 8;
  union { uint4 q; unsigned short h[8]; } a, b;
  a.q = *(const uint4*)(Yb + (size_t)p0 * D + c);
  b.q = *(const uint4*)(Yb + (size_t)p1 * D + c);
  float res[8];
#pragma unroll
  for (int j = 0; j < 8; ++j)
    res[j] = w0 * bf2f(a.h[j]) + w1 * bf2f(b.h[j]);
  float* o = out + (size_t)t * D + c;
  float4 v0 = {res[0], res[1], res[2], res[3]};
  float4 v1 = {res[4], res[5], res[6], res[7]};
  ((float4*)o)[0] = v0;
  ((float4*)o)[1] = v1;
}

// -----------------------------------------------------------------------------
extern "C" void kernel_launch(void* const* d_in, const int* in_sizes, int n_in,
                              void* d_out, int out_size, void* d_ws, size_t ws_size,
                              hipStream_t stream) {
  (void)in_sizes; (void)n_in; (void)out_size; (void)ws_size;
  const float* x      = (const float*)d_in[0];   // [4,2048,2048]
  const float* gate_w = (const float*)d_in[1];   // [16,2048]
  const float* wg     = (const float*)d_in[2];   // [16,2048,1408]
  const float* wu     = (const float*)d_in[3];   // [16,2048,1408]
  const float* wd     = (const float*)d_in[4];   // [16,1408,2048]
  float* out = (float*)d_out;

  char* p = (char*)d_ws;
  unsigned short* xb = (unsigned short*)p; p += (size_t)NTOK * D * 2;   // 32 MB
  unsigned short* Hb = (unsigned short*)p; p += (size_t)PMAX * I * 2;   // 50 MB
  unsigned short* Yb = (unsigned short*)p; p += (size_t)PMAX * D * 2;   // 72 MB
  unsigned short* zeropad = (unsigned short*)p; p += (size_t)D * 2;     // 4 KB
  int*   sel        = (int*)p;   p += (size_t)NTOK * 2 * sizeof(int);
  float* rwv        = (float*)p; p += (size_t)NTOK * 2 * sizeof(float);
  int*   pairidx    = (int*)p;   p += (size_t)NTOK * 2 * sizeof(int);
  int*   pair_token = (int*)p;   p += (size_t)PMAX * sizeof(int);
  int*   counts     = (int*)p;   p += 64;
  int*   cursor     = (int*)p;   p += 64;
  int*   tile_expert = (int*)p;  p += (size_t)NTILES * sizeof(int);
  int*   tile_base   = (int*)p;  p += (size_t)NTILES * sizeof(int);

  moe_init<<<(PMAX + 255) / 256, 256, 0, stream>>>(pair_token, counts, tile_expert,
                                                   tile_base, zeropad);
  moe_router<<<NTOK / 256, 256, 0, stream>>>(x, gate_w, xb, sel, rwv, counts);
  moe_setup<<<1, 32, 0, stream>>>(counts, cursor, tile_expert, tile_base);
  moe_assign<<<NTOK / 256, 256, 0, stream>>>(sel, cursor, pair_token, pairidx);
  moe_gateup<<<dim3(NTILES, I / TN), 256, 0, stream>>>(xb, wg, wu, pair_token,
                                                       tile_expert, tile_base,
                                                       zeropad, Hb);
  moe_down<<<dim3(NTILES, D / TN), 256, 0, stream>>>(Hb, wd, tile_expert, tile_base, Yb);
  moe_combine<<<NTOK, 256, 0, stream>>>(Yb, pairidx, rwv, out);
}
